// FOL_single_input_32487132627310
// MI455X (gfx1250) — matrix operand order repaired and checked by reading, not compiled, not run on hardware
//
#include <hip/hip_runtime.h>
#include <hip/hip_bf16.h>
#include <math.h>

// Problem constants (match reference)
#define B_   1024
#define S_   10
#define T_   10
#define E_   512
#define BH_  512
#define DH_  512
#define DEC_ 1024
#define PI_  512

typedef __bf16 bf16;
typedef __attribute__((ext_vector_type(16))) __bf16 bf16x16;
typedef __attribute__((ext_vector_type(8)))  __bf16 bf16x8;
typedef __attribute__((ext_vector_type(8)))  float  f32x8;
typedef __attribute__((ext_vector_type(4)))  unsigned int u32x4;
typedef __attribute__((ext_vector_type(8)))  int i32x8;
typedef __attribute__((ext_vector_type(4)))  int i32x4;

#define HAVE_TDM __has_builtin(__builtin_amdgcn_tensor_load_to_lds)

// ---------------- zero fill (16B granules) ----------------
__global__ void zero4_kernel(uint4* __restrict__ p, long n) {
  long i = (long)blockIdx.x * blockDim.x + threadIdx.x;
  if (i < n) p[i] = make_uint4(0u, 0u, 0u, 0u);
}

// ------- weight convert+transpose: W (N x K) f32 -> Wt (K x N) bf16 -------
__global__ void wt_kernel(const float* __restrict__ W, bf16* __restrict__ Wt,
                          int N, int K) {
  long i = (long)blockIdx.x * blockDim.x + threadIdx.x;   // i = k*N + n
  long total = (long)N * K;
  if (i >= total) return;
  int n = (int)(i % N);
  int k = (int)(i / N);
  Wt[i] = (bf16)W[(long)n * K + k];
}

// ------- embedding: out[(s*B+b)*E + e] = relu(x[b,s,off..off+3].w[e,:]+bias[e]) -------
__global__ void embed_kernel(const float* __restrict__ x, int off,
                             const float* __restrict__ w,
                             const float* __restrict__ bias,
                             bf16* __restrict__ out) {
  long i = (long)blockIdx.x * blockDim.x + threadIdx.x;   // over S_*B_*E_
  if (i >= (long)S_ * B_ * E_) return;
  int e  = (int)(i % E_);
  long sb = i / E_;
  int b  = (int)(sb % B_);
  int s  = (int)(sb / B_);
  const float* xp = x + ((long)b * S_ + s) * 8 + off;
  const float* wr = w + (long)e * 4;
  float acc = bias[e] + xp[0]*wr[0] + xp[1]*wr[1] + xp[2]*wr[2] + xp[3]*wr[3];
  out[i] = (bf16)fmaxf(acc, 0.0f);
}

// ---------------- TDM: 2D tile load Global -> LDS (ISA 8.3/8.4 D# layout) ----------------
// Loads tile_h rows x tile_w elements (bf16, data_size=2B) from a row-major
// tensor with row stride `row_stride` (elements), into LDS at byte offset
// `lds_byte`, inserting `pada+1` dwords of pad every 2^(padi+1) dwords
// (hardware-applied LDS pitch padding).
__device__ __forceinline__ void tdm_load_tile(unsigned lds_byte, const bf16* g,
                                              long row_stride,
                                              int tile_w, int tile_h,
                                              unsigned tensor_d0, unsigned tensor_d1,
                                              int padi, int pada) {
#if HAVE_TDM
  size_t ga = (size_t)g;
  u32x4 g0;
  g0[0] = 1u;                                               // count=1, user D#
  g0[1] = lds_byte;                                         // lds_addr
  g0[2] = (unsigned)ga;                                     // global_addr[31:0]
  g0[3] = ((unsigned)(ga >> 32) & 0x01FFFFFFu) | (2u << 30);// [56:32] | type=2
  i32x8 g1;
  g1[0] = (int)((1u << 16)                                  // data_size: 2 bytes
              | (1u << 20)                                  // pad_enable
              | ((unsigned)padi << 22)                      // pad_interval
              | ((unsigned)pada << 25));                    // pad_amount
  g1[1] = (int)((tensor_d0 & 0xFFFFu) << 16);               // tensor_dim0[15:0]
  g1[2] = (int)((tensor_d0 >> 16) | ((tensor_d1 & 0xFFFFu) << 16));
  g1[3] = (int)((tensor_d1 >> 16) | ((unsigned)tile_w << 16)); // tile_dim0
  g1[4] = (int)(unsigned)tile_h;                            // tile_dim1 (dim2=0)
  g1[5] = (int)(unsigned)(row_stride & 0xFFFFFFFFL);        // dim0_stride lo
  g1[6] = (int)(unsigned)((row_stride >> 32) & 0xFFFFL);    // dim0_stride hi
  g1[7] = 0;
  i32x4 z4 = {0, 0, 0, 0};
#if defined(__clang_major__) && (__clang_major__ >= 23)
  // clang-23 / therock lane: 6-arg form (extra i32x8 before cpol)
  i32x8 z8 = {0, 0, 0, 0, 0, 0, 0, 0};
  __builtin_amdgcn_tensor_load_to_lds(g0, g1, z4, z4, z8, 0);
#else
  // ROCm 7.2 / clang-22 lane: 5-arg form
  __builtin_amdgcn_tensor_load_to_lds(g0, g1, z4, z4, 0);
#endif
#else
  (void)lds_byte; (void)g; (void)row_stride; (void)tile_w; (void)tile_h;
  (void)tensor_d0; (void)tensor_d1; (void)padi; (void)pada;
#endif
}

// ---------------- core WMMA GEMM ----------------
// C(M,N) = A(MxK bf16, row-major) * Wt(KxN bf16, row-major) + bias[N]
// act: 0 = none, 1 = relu.  Writes Cf (f32) and/or Cbf (bf16) if non-null.
// Block tile 64(M) x 128(N), 8 waves (2x4), 32x32 per wave, K step 32.
// TDM double-buffered staging: wave 0 issues two tensor_load_to_lds per step,
// TENSORcnt-throttled one step ahead of the WMMA pipeline.
#define APITCH 40   // 80 B/row via TDM pad (4 dwords per 16 dwords)
#define BPITCH 136  // 272 B/row via TDM pad (4 dwords per 64 dwords)
__global__ __launch_bounds__(256)
void gemm_kernel(const bf16* __restrict__ A, const bf16* __restrict__ Wt,
                 const float* __restrict__ bias,
                 float* __restrict__ Cf, bf16* __restrict__ Cbf,
                 int M, int N, int K, int act) {
  __shared__ bf16 As[2][64 * APITCH];
  __shared__ bf16 Bs[2][32 * BPITCH];
  const int tid  = threadIdx.x;
  const int lane = tid & 31;
  const int wave = tid >> 5;
  const int wm   = wave >> 2;            // 0..1
  const int wn   = wave & 3;             // 0..3
  const int m0   = blockIdx.y * 64;
  const int n0   = blockIdx.x * 128;

  f32x8 acc[2][2];
  #pragma unroll
  for (int i = 0; i < 2; ++i)
    #pragma unroll
    for (int j = 0; j < 2; ++j)
      acc[i][j] = f32x8{0.f,0.f,0.f,0.f,0.f,0.f,0.f,0.f};

#if HAVE_TDM
  // Scalar-branch issuer: readfirstlane makes the condition wave-uniform so
  // only wave 0 reaches the TDM issue (TDM ignores EXEC).
  const bool issuer = (__builtin_amdgcn_readfirstlane((int)wave) == 0);
  const int nt = K >> 5;
  if (issuer) {
    tdm_load_tile((unsigned)(size_t)&As[0][0], &A[(long)m0 * K], K,
                  32, 64, (unsigned)K, (unsigned)M, /*padi=*/3, /*pada=*/3);
    tdm_load_tile((unsigned)(size_t)&Bs[0][0], &Wt[(long)0 * N + n0], N,
                  128, 32, (unsigned)N, (unsigned)K, /*padi=*/5, /*pada=*/3);
  }
  for (int it = 0; it < nt; ++it) {
    const int buf = it & 1;
    const bool more = (it + 1 < nt);
    if (issuer && more) {
      const int kn = (it + 1) << 5;
      tdm_load_tile((unsigned)(size_t)&As[buf ^ 1][0], &A[(long)m0 * K + kn], K,
                    32, 64, (unsigned)K, (unsigned)M, 3, 3);
      tdm_load_tile((unsigned)(size_t)&Bs[buf ^ 1][0], &Wt[(long)kn * N + n0], N,
                    128, 32, (unsigned)N, (unsigned)K, 5, 3);
    }
    if (more) __builtin_amdgcn_s_wait_tensorcnt(2);  // tile `it` landed
    else      __builtin_amdgcn_s_wait_tensorcnt(0);
    __syncthreads();
#else
  const int arow = tid >> 2, achk = tid & 3;   // A stage: 64 rows x 4 x 16B
  const int brow = tid >> 3, bchk = tid & 7;   // B stage: 32 rows x 8 x 32B
  const int nt = K >> 5;
  for (int it = 0; it < nt; ++it) {
    const int buf = it & 1;
    const int k0 = it << 5;
    *(bf16x8*)&As[buf][arow * APITCH + achk * 8] =
        *(const bf16x8*)&A[(long)(m0 + arow) * K + k0 + achk * 8];
    {
      const bf16* src = &Wt[(long)(k0 + brow) * N + n0 + bchk * 16];
      *(bf16x8*)&Bs[buf][brow * BPITCH + bchk * 16    ] = *(const bf16x8*)(src);
      *(bf16x8*)&Bs[buf][brow * BPITCH + bchk * 16 + 8] = *(const bf16x8*)(src + 8);
    }
    __syncthreads();
#endif

    // Fragments per ISA 7.12.2 layouts.
    // A (16x32 bf16): lane row = lane&15; elems 0-7 -> K = kb..kb+7,
    // elems 8-15 -> K = 16+kb..16+kb+7, kb = 8*(lane>>4).
    const int ar = lane & 15;
    const int kb = (lane >> 4) << 3;
    bf16x16 af[2], bfr[2];
    #pragma unroll
    for (int i = 0; i < 2; ++i) {
      int r = wm * 32 + i * 16 + ar;
      bf16x8 lo = *(const bf16x8*)&As[buf][r * APITCH + kb];
      bf16x8 hi = *(const bf16x8*)&As[buf][r * APITCH + 16 + kb];
      #pragma unroll
      for (int e = 0; e < 8; ++e) { af[i][e] = lo[e]; af[i][e + 8] = hi[e]; }
    }
    // B (32x16 bf16): lane col = lane&15; elems 0-7 -> K = kb..kb+7,
    // elems 8-15 -> K = 16+kb..16+kb+7 (same as A).
    #pragma unroll
    for (int j = 0; j < 2; ++j) {
      int c = wn * 32 + j * 16 + ar;
      #pragma unroll
      for (int e = 0; e < 8; ++e) {
        bfr[j][e]     = Bs[buf][(kb + e) * BPITCH + c];
        bfr[j][e + 8] = Bs[buf][(16 + kb + e) * BPITCH + c];
      }
    }

    #pragma unroll
    for (int i = 0; i < 2; ++i)
      #pragma unroll
      for (int j = 0; j < 2; ++j)
        acc[i][j] = __builtin_amdgcn_wmma_f32_16x16x32_bf16(
            false, af[i], false, bfr[j], (short)0, acc[i][j], false, false);

    __syncthreads();   // all waves done reading `buf` -> safe to refill next iter
  }

  // epilogue: C layout — VGPR v: m = v + 8*(lane>>4), n = lane&15
  const int cn = lane & 15;
  const int ch = (lane >> 4) << 3;
  #pragma unroll
  for (int j = 0; j < 2; ++j) {
    int n = n0 + wn * 32 + j * 16 + cn;
    float bv = bias ? bias[n] : 0.0f;
    #pragma unroll
    for (int i = 0; i < 2; ++i) {
      int mb = m0 + wm * 32 + i * 16 + ch;
      #pragma unroll
      for (int v = 0; v < 8; ++v) {
        float xv = acc[i][j][v] + bv;
        if (act == 1) xv = fmaxf(xv, 0.0f);
        long idx = (long)(mb + v) * N + n;
        if (Cf)  Cf[idx]  = xv;
        if (Cbf) Cbf[idx] = (bf16)xv;
      }
    }
  }
}

// ---------------- GRU gates (fp32), writes new state + bf16 copy ----------------
__global__ void gru_gate_kernel(const float* __restrict__ gi,
                                const float* __restrict__ gh,
                                float* __restrict__ h, bf16* __restrict__ hbf,
                                float* __restrict__ hcopy, long copyRowStride,
                                int H) {
  long i = (long)blockIdx.x * blockDim.x + threadIdx.x;  // over B_*H
  if (i >= (long)B_ * H) return;
  int b = (int)(i / H);
  int d = (int)(i % H);
  const float* gib = gi + (long)b * 3 * H;
  const float* ghb = gh + (long)b * 3 * H;
  float r = 1.0f / (1.0f + __expf(-(gib[d]         + ghb[d])));
  float z = 1.0f / (1.0f + __expf(-(gib[H + d]     + ghb[H + d])));
  float n = tanhf(gib[2 * H + d] + r * ghb[2 * H + d]);
  float hv = (1.0f - z) * n + z * h[i];
  h[i]   = hv;
  hbf[i] = (bf16)hv;
  if (hcopy) hcopy[(long)b * copyRowStride + d] = hv;
}

// ---------------- concat box_h|dm_h -> decoder h (f32 + bf16) ----------------
__global__ void concat_kernel(const float* __restrict__ a,
                              const float* __restrict__ b,
                              float* __restrict__ hf, bf16* __restrict__ hbf) {
  long i = (long)blockIdx.x * blockDim.x + threadIdx.x;  // over B_*DEC_
  if (i >= (long)B_ * DEC_) return;
  int row = (int)(i >> 10);
  int d   = (int)(i & 1023);
  float v = (d < BH_) ? a[(long)row * BH_ + d] : b[(long)row * DH_ + (d - BH_)];
  hf[i]  = v;
  hbf[i] = (bf16)v;
}

// ---------------- 4-wide tanh head (fp32, wave-shuffle reduce) ----------------
__global__ __launch_bounds__(128)
void head_kernel(const float* __restrict__ h, const float* __restrict__ w,
                 const float* __restrict__ b4, float* __restrict__ preds,
                 int s, int t) {
  int row  = blockIdx.x;          // batch index
  int p    = threadIdx.x >> 5;    // output 0..3 (one wave each)
  int lane = threadIdx.x & 31;
  const float* hr = h + (long)row * DEC_;
  const float* wr = w + (long)p * DEC_;
  float acc = 0.0f;
  for (int k = lane; k < DEC_; k += 32) acc += hr[k] * wr[k];
  #pragma unroll
  for (int off = 16; off > 0; off >>= 1) acc += __shfl_down(acc, off, 32);
  if (lane == 0)
    preds[(((long)row * S_ + s) * T_ + t) * 4 + p] = tanhf(acc + b4[p]);
}

// ======================== host driver ========================
extern "C" void kernel_launch(void* const* d_in, const int* in_sizes, int n_in,
                              void* d_out, int out_size, void* d_ws, size_t ws_size,
                              hipStream_t stream) {
  (void)in_sizes; (void)n_in; (void)out_size; (void)ws_size;
  const float* x        = (const float*)d_in[0];
  const float* bew      = (const float*)d_in[1];
  const float* beb      = (const float*)d_in[2];
  const float* dew      = (const float*)d_in[3];
  const float* deb      = (const float*)d_in[4];
  const float* box_wih  = (const float*)d_in[5];
  const float* box_whh  = (const float*)d_in[6];
  const float* box_bih  = (const float*)d_in[7];
  const float* box_bhh  = (const float*)d_in[8];
  const float* dm_wih   = (const float*)d_in[9];
  const float* dm_whh   = (const float*)d_in[10];
  const float* dm_bih   = (const float*)d_in[11];
  const float* dm_bhh   = (const float*)d_in[12];
  const float* dec_wih  = (const float*)d_in[13];
  const float* dec_whh  = (const float*)d_in[14];
  const float* dec_bih  = (const float*)d_in[15];
  const float* dec_bhh  = (const float*)d_in[16];
  const float* h2pi_w   = (const float*)d_in[17];
  const float* h2pi_b   = (const float*)d_in[18];
  const float* h2p_w    = (const float*)d_in[19];
  const float* h2p_b    = (const float*)d_in[20];

  float* preds = (float*)d_out;                         // (B,S,T,4)
  float* dech  = preds + (size_t)B_ * S_ * T_ * 4;      // (B,S,T,DEC)

  // ---- bump allocator over workspace ----
  char* wsp = (char*)d_ws;
  auto alloc = [&](size_t bytes) -> void* {
    void* r = (void*)wsp;
    wsp += (bytes + 255) & ~(size_t)255;
    return r;
  };
  bf16* Wt_box_wih = (bf16*)alloc((size_t)512  * 1536 * 2);
  bf16* Wt_box_whh = (bf16*)alloc((size_t)512  * 1536 * 2);
  bf16* Wt_dm_wih  = (bf16*)alloc((size_t)512  * 1536 * 2);
  bf16* Wt_dm_whh  = (bf16*)alloc((size_t)512  * 1536 * 2);
  bf16* Wt_dec_wih = (bf16*)alloc((size_t)512  * 3072 * 2);
  bf16* Wt_dec_whh = (bf16*)alloc((size_t)1024 * 3072 * 2);
  bf16* Wt_h2pi    = (bf16*)alloc((size_t)1024 * 512  * 2);
  bf16* emb_box    = (bf16*)alloc((size_t)S_ * B_ * E_ * 2);
  bf16* emb_dm     = (bf16*)alloc((size_t)S_ * B_ * E_ * 2);
  float* h_box_f   = (float*)alloc((size_t)B_ * BH_  * 4);
  bf16*  h_box_b   = (bf16*) alloc((size_t)B_ * BH_  * 2);
  float* h_dm_f    = (float*)alloc((size_t)B_ * DH_  * 4);
  bf16*  h_dm_b    = (bf16*) alloc((size_t)B_ * DH_  * 2);
  float* h_dec_f   = (float*)alloc((size_t)B_ * DEC_ * 4);
  bf16*  h_dec_b   = (bf16*) alloc((size_t)B_ * DEC_ * 2);
  bf16*  pi_b      = (bf16*) alloc((size_t)B_ * PI_  * 2);
  float* gi_buf    = (float*)alloc((size_t)B_ * 3072 * 4);
  float* gh_buf    = (float*)alloc((size_t)B_ * 3072 * 4);

  auto zero = [&](void* p, size_t bytes) {
    long n = (long)(bytes / 16);
    zero4_kernel<<<(n + 255) / 256, 256, 0, stream>>>((uint4*)p, n);
  };
  auto wconv = [&](const float* W, bf16* Wt, int N, int K) {
    long total = (long)N * K;
    wt_kernel<<<(total + 255) / 256, 256, 0, stream>>>(W, Wt, N, K);
  };
  auto gemm = [&](const bf16* A, const bf16* Wt, const float* bias,
                  float* Cf, bf16* Cbf, int M, int N, int K, int act) {
    gemm_kernel<<<dim3(N / 128, M / 64), 256, 0, stream>>>(
        A, Wt, bias, Cf, Cbf, M, N, K, act);
  };

  // ---- one-time (per launch) prep ----
  wconv(box_wih, Wt_box_wih, 1536, 512);
  wconv(box_whh, Wt_box_whh, 1536, 512);
  wconv(dm_wih,  Wt_dm_wih,  1536, 512);
  wconv(dm_whh,  Wt_dm_whh,  1536, 512);
  wconv(dec_wih, Wt_dec_wih, 3072, 512);
  wconv(dec_whh, Wt_dec_whh, 3072, 1024);
  wconv(h2pi_w,  Wt_h2pi,    512,  1024);

  {
    long n = (long)S_ * B_ * E_;
    embed_kernel<<<(n + 255) / 256, 256, 0, stream>>>(x, 0, bew, beb, emb_box);
    embed_kernel<<<(n + 255) / 256, 256, 0, stream>>>(x, 4, dew, deb, emb_dm);
  }
  zero(h_box_f, (size_t)B_ * BH_ * 4);
  zero(h_box_b, (size_t)B_ * BH_ * 2);
  zero(h_dm_f,  (size_t)B_ * DH_ * 4);
  zero(h_dm_b,  (size_t)B_ * DH_ * 2);

  // ---- sequential recurrence ----
  for (int s = 0; s < S_; ++s) {
    // box GRU
    gemm(emb_box + (size_t)s * B_ * E_, Wt_box_wih, box_bih, gi_buf, nullptr,
         B_, 1536, 512, 0);
    gemm(h_box_b, Wt_box_whh, box_bhh, gh_buf, nullptr, B_, 1536, 512, 0);
    gru_gate_kernel<<<(B_ * BH_) / 256, 256, 0, stream>>>(
        gi_buf, gh_buf, h_box_f, h_box_b, nullptr, 0, BH_);
    // dm GRU
    gemm(emb_dm + (size_t)s * B_ * E_, Wt_dm_wih, dm_bih, gi_buf, nullptr,
         B_, 1536, 512, 0);
    gemm(h_dm_b, Wt_dm_whh, dm_bhh, gh_buf, nullptr, B_, 1536, 512, 0);
    gru_gate_kernel<<<(B_ * DH_) / 256, 256, 0, stream>>>(
        gi_buf, gh_buf, h_dm_f, h_dm_b, nullptr, 0, DH_);
    // decoder init: h = concat(box_h, dm_h); pi = 0
    concat_kernel<<<(B_ * DEC_) / 256, 256, 0, stream>>>(
        h_box_f, h_dm_f, h_dec_f, h_dec_b);
    zero(pi_b, (size_t)B_ * PI_ * 2);

    for (int t = 0; t < T_; ++t) {
      gemm(pi_b,    Wt_dec_wih, dec_bih, gi_buf, nullptr, B_, 3072, 512, 0);
      gemm(h_dec_b, Wt_dec_whh, dec_bhh, gh_buf, nullptr, B_, 3072, 1024, 0);
      gru_gate_kernel<<<(B_ * DEC_) / 256, 256, 0, stream>>>(
          gi_buf, gh_buf, h_dec_f, h_dec_b,
          dech + (size_t)(s * T_ + t) * DEC_, (long)S_ * T_ * DEC_, DEC_);
      // pi = relu(h @ h2pi^T + b)  (bf16 out, feeds next step's GEMM)
      gemm(h_dec_b, Wt_h2pi, h2pi_b, nullptr, pi_b, B_, 512, 1024, 1);
      // out = tanh(h @ h2p^T + b)  (fp32 path)
      head_kernel<<<B_, 128, 0, stream>>>(h_dec_f, h2p_w, h2p_b, preds, s, t);
    }
  }
}
